// PopulationODE_32753420599423
// MI455X (gfx1250) — matrix operand order repaired and checked by reading, not compiled, not run on hardware
//
#include <hip/hip_runtime.h>

typedef __attribute__((ext_vector_type(16))) _Float16 v16h;
typedef __attribute__((ext_vector_type(8)))  _Float16 v8h;
typedef __attribute__((ext_vector_type(8)))  float    v8f;

#define WAVE_FENCE() __builtin_amdgcn_fence(__ATOMIC_ACQ_REL, "wavefront")

__device__ __forceinline__ v8f wmma16(v16h a, v16h b, v8f c) {
  // D = A(16x32 f16) * B(32x16 f16) + C(16x16 f32)
  return __builtin_amdgcn_wmma_f32_16x16x32_f16(false, a, false, b, (short)0, c,
                                                false, false);
}

// ---- branchless transcendentals on the TRANS pipe (no OCML EXEC juggling) ----
__device__ __forceinline__ float fast_tanh(float x) {
#if __has_builtin(__builtin_amdgcn_tanhf)
  return __builtin_amdgcn_tanhf(x);            // v_tanh_f32
#elif __has_builtin(__builtin_amdgcn_tanh_f32)
  return __builtin_amdgcn_tanh_f32(x);         // v_tanh_f32
#else
  const float e = __builtin_amdgcn_exp2f(2.8853900817779268f * x); // 2*log2(e)*x
  return 1.0f - 2.0f * __builtin_amdgcn_rcpf(e + 1.0f);
#endif
}

__device__ __forceinline__ float fast_gauss(float g) {
  // exp(-g^2) = exp2(-log2(e) * g^2)  -> single v_exp_f32
  return __builtin_amdgcn_exp2f(-1.4426950408889634f * (g * g));
}

// ---- 16-lane (DPP-row) sum reduction on the VALU pipe: 4 x v_add_f32_dpp ----
template <int CTRL>
__device__ __forceinline__ float dpp_xadd(float x) {
  const int xi = __builtin_bit_cast(int, x);
  const int sh = __builtin_amdgcn_update_dpp(0, xi, CTRL, 0xF, 0xF, true);
  return x + __builtin_bit_cast(float, sh);
}

__device__ __forceinline__ float row16_sum(float v) {
  v = dpp_xadd<0xB1>(v);    // quad_perm [1,0,3,2]  (xor 1)
  v = dpp_xadd<0x4E>(v);    // quad_perm [2,3,0,1]  (xor 2)
  v = dpp_xadd<0x141>(v);   // row_half_mirror: quad <-> adjacent quad
  v = dpp_xadd<0x140>(v);   // row_mirror: across the 16-lane row
  return v;
}

constexpr int   TSTEPS = 256;
constexpr int   BROWS  = 65536;
constexpr float EPS    = 1e-5f;

__global__ __launch_bounds__(256)
void pop_ode_kernel(const float* __restrict__ s_grid,
                    const float* __restrict__ y0,
                    const float* __restrict__ W1, const float* __restrict__ b1,
                    const float* __restrict__ W2, const float* __restrict__ b2,
                    const float* __restrict__ G1, const float* __restrict__ gb1,
                    const float* __restrict__ l1g, const float* __restrict__ l1b,
                    const float* __restrict__ G2, const float* __restrict__ gb2,
                    const float* __restrict__ l2g, const float* __restrict__ l2b,
                    const float* __restrict__ G3, const float* __restrict__ gb3,
                    float* __restrict__ out)
{
  // per-wave staging: activation tile (C-layout -> A-layout transpose), staged y
  __shared__ __align__(16) _Float16 sACT[8][16][32];   // 8 KB
  __shared__ __align__(16) float    sYS[8][16][4];     // 2 KB

  const int lane  = threadIdx.x & 31;
  const int wv    = threadIdx.x >> 5;
  const int col16 = lane & 15;       // C layout: my column within a 16-col tile
  const int hi    = lane >> 4;       // C layout: rows i+8*hi ; A layout: K half
  const int grow0 = blockIdx.x * 128 + wv * 16;  // first global row of tile

  // ---- weight B-fragments (32x16 f16: lane%16 = column, element j = K in table order) ----
  v16h w1b0, w1b1, g1b0, g1b1, g2b0, g2b1, w2b, g3b;
  for (int j = 0; j < 16; ++j) {
    const int k = (j < 8) ? (8 * hi + j) : (16 + 8 * hi + (j - 8));
    w1b0[j] = (_Float16)((k < 4) ? W1[k * 32 + col16]      : 0.f);
    w1b1[j] = (_Float16)((k < 4) ? W1[k * 32 + 16 + col16] : 0.f);
    g1b0[j] = (_Float16)((k < 4) ? G1[k * 32 + col16]      : 0.f);
    g1b1[j] = (_Float16)((k < 4) ? G1[k * 32 + 16 + col16] : 0.f);
    g2b0[j] = (_Float16)G2[k * 32 + col16];
    g2b1[j] = (_Float16)G2[k * 32 + 16 + col16];
    w2b[j]  = (_Float16)((col16 < 4) ? W2[k * 4 + col16] : 0.f);
    g3b[j]  = (_Float16)((col16 < 4) ? G3[k * 4 + col16] : 0.f);
  }

  // ---- column-indexed parameters (C layout: my column = col16 / col16+16) ----
  const float b1c0 = b1[col16],  b1c1 = b1[col16 + 16];
  const float q1c0 = gb1[col16], q1c1 = gb1[col16 + 16];
  const float q2c0 = gb2[col16], q2c1 = gb2[col16 + 16];
  const float g1g0 = l1g[col16], g1g1 = l1g[col16 + 16];
  const float g1b_ = l1b[col16], g1b1_ = l1b[col16 + 16];
  const float g2g0 = l2g[col16], g2g1 = l2g[col16 + 16];
  const float g2b_ = l2b[col16], g2b1_ = l2b[col16 + 16];
  const float b2c  = (col16 < 4) ? b2[col16]  : 0.f;
  const float q3c  = (col16 < 4) ? gb3[col16] : 0.f;

  // ---- y state in C layout: yreg[i] = y[row i+8*hi][col16] (valid col16<4) ----
  float yreg[8];
  for (int i = 0; i < 8; ++i) {
    const int r = grow0 + i + 8 * hi;
    yreg[i] = (col16 < 4) ? y0[(size_t)r * 4 + col16] : 0.f;
  }

  // stage y (C-layout regs) -> LDS row-major 16x4
  auto write_ys = [&](const float* v) {
    if (col16 < 4) {
      for (int i = 0; i < 8; ++i) sYS[wv][i + 8 * hi][col16] = v[i];
    }
  };

  // A fragment (16x32 f16) from the row-major LDS activation tile
  auto load_afrag = [&]() -> v16h {
    const v8h* p = (const v8h*)&sACT[wv][col16][8 * hi];
    v8h lo = p[0];          // K = 8*hi .. 8*hi+7
    v8h hh = p[2];          // K = 16+8*hi .. 16+8*hi+7
    return __builtin_shufflevector(lo, hh, 0, 1, 2, 3, 4, 5, 6, 7,
                                   8, 9, 10, 11, 12, 13, 14, 15);
  };

  // LayerNorm over a 32-wide row held as (a0, a1) across a 16-lane group
  auto lnorm = [&](float* a0, float* a1, float g0, float g1, float c0, float c1) {
    for (int i = 0; i < 8; ++i) {
      const float s = row16_sum(a0[i] + a1[i]);
      const float q = row16_sum(a0[i] * a0[i] + a1[i] * a1[i]);
      const float mean = s * (1.f / 32.f);
      const float var  = q * (1.f / 32.f) - mean * mean;
      const float rs   = __builtin_amdgcn_rsqf(var + EPS);   // v_rsq_f32
      a0[i] = (a0[i] - mean) * rs * g0 + c0;
      a1[i] = (a1[i] - mean) * rs * g1 + c1;
    }
  };

  // f(y_staged) -> k in C layout (cols 0..3 of tile 0 valid)
  auto feval = [&](float* kc) {
    // A fragment of staged y: K=0..3 live in lanes hi==0, everything else 0
    v16h ay = {};
    if (hi == 0) {
      const float* p = sYS[wv][col16];
      ay[0] = (_Float16)p[0]; ay[1] = (_Float16)p[1];
      ay[2] = (_Float16)p[2]; ay[3] = (_Float16)p[3];
    }

    // ---- magnitude path: h = tanh(y@W1 + b1); mag = h@W2 + b2 ----
    v8f c0 = {}, c1 = {};
    c0 = wmma16(ay, w1b0, c0);
    c1 = wmma16(ay, w1b1, c1);
    WAVE_FENCE();
    for (int i = 0; i < 8; ++i) {
      sACT[wv][i + 8 * hi][col16]      = (_Float16)fast_tanh(c0[i] + b1c0);
      sACT[wv][i + 8 * hi][col16 + 16] = (_Float16)fast_tanh(c1[i] + b1c1);
    }
    WAVE_FENCE();
    v16h ah = load_afrag();
    v8f mc = {};
    mc = wmma16(ah, w2b, mc);          // mag (cols 0..3) before +b2

    // ---- gating path ----
    v8f d0 = {}, d1 = {};
    d0 = wmma16(ay, g1b0, d0);
    d1 = wmma16(ay, g1b1, d1);
    float t0[8], t1[8];
    for (int i = 0; i < 8; ++i) {
      t0[i] = fast_tanh(d0[i] + q1c0);
      t1[i] = fast_tanh(d1[i] + q1c1);
    }
    lnorm(t0, t1, g1g0, g1g1, g1b_, g1b1_);
    WAVE_FENCE();
    for (int i = 0; i < 8; ++i) {
      sACT[wv][i + 8 * hi][col16]      = (_Float16)t0[i];
      sACT[wv][i + 8 * hi][col16 + 16] = (_Float16)t1[i];
    }
    WAVE_FENCE();
    v16h at = load_afrag();
    d0 = v8f{}; d1 = v8f{};
    d0 = wmma16(at, g2b0, d0);
    d1 = wmma16(at, g2b1, d1);
    for (int i = 0; i < 8; ++i) {
      t0[i] = fast_tanh(d0[i] + q2c0);
      t1[i] = fast_tanh(d1[i] + q2c1);
    }
    lnorm(t0, t1, g2g0, g2g1, g2b_, g2b1_);
    WAVE_FENCE();
    for (int i = 0; i < 8; ++i) {
      sACT[wv][i + 8 * hi][col16]      = (_Float16)t0[i];
      sACT[wv][i + 8 * hi][col16 + 16] = (_Float16)t1[i];
    }
    WAVE_FENCE();
    v16h at2 = load_afrag();
    v8f gc = {};
    gc = wmma16(at2, g3b, gc);

    for (int i = 0; i < 8; ++i) {
      kc[i] = (mc[i] + b2c) * fast_gauss(gc[i] + q3c);   // mag * gaussian gate
    }
  };

  // coalesced store of this wave's 16x4 tile for timestep t
  auto store_out = [&](int t) {
    const float2 v = ((const float2*)&sYS[wv][0][0])[lane];
    ((float2*)(out + ((size_t)t * BROWS + grow0) * 4))[lane] = v;
  };

  // ---- t = 0: emit y0 ----
  write_ys(yreg);
  WAVE_FENCE();
  store_out(0);

  // ---- RK4 time loop ----
  float k1[8], k2[8], k3[8], k4[8], tmp[8];
  for (int t = 1; t < TSTEPS; ++t) {
    const float h = s_grid[t] - s_grid[t - 1];

    feval(k1);
    for (int i = 0; i < 8; ++i) tmp[i] = yreg[i] + 0.5f * h * k1[i];
    WAVE_FENCE(); write_ys(tmp); WAVE_FENCE();

    feval(k2);
    for (int i = 0; i < 8; ++i) tmp[i] = yreg[i] + 0.5f * h * k2[i];
    WAVE_FENCE(); write_ys(tmp); WAVE_FENCE();

    feval(k3);
    for (int i = 0; i < 8; ++i) tmp[i] = yreg[i] + h * k3[i];
    WAVE_FENCE(); write_ys(tmp); WAVE_FENCE();

    feval(k4);
    const float h6 = h * (1.f / 6.f);
    for (int i = 0; i < 8; ++i)
      yreg[i] += h6 * (k1[i] + 2.f * (k2[i] + k3[i]) + k4[i]);

    WAVE_FENCE(); write_ys(yreg); WAVE_FENCE();
    store_out(t);
  }
}

extern "C" void kernel_launch(void* const* d_in, const int* in_sizes, int n_in,
                              void* d_out, int out_size, void* d_ws, size_t ws_size,
                              hipStream_t stream) {
  const float* s_grid = (const float*)d_in[0];
  const float* y0     = (const float*)d_in[1];
  const float* W1     = (const float*)d_in[2];
  const float* b1     = (const float*)d_in[3];
  const float* W2     = (const float*)d_in[4];
  const float* b2     = (const float*)d_in[5];
  const float* G1     = (const float*)d_in[6];
  const float* gb1    = (const float*)d_in[7];
  const float* l1g    = (const float*)d_in[8];
  const float* l1b    = (const float*)d_in[9];
  const float* G2     = (const float*)d_in[10];
  const float* gb2    = (const float*)d_in[11];
  const float* l2g    = (const float*)d_in[12];
  const float* l2b    = (const float*)d_in[13];
  const float* G3     = (const float*)d_in[14];
  const float* gb3    = (const float*)d_in[15];
  float* out = (float*)d_out;

  dim3 grid(BROWS / 128);   // 512 blocks, 128 rows per block
  dim3 block(256);          // 8 waves x 16 rows
  hipLaunchKernelGGL(pop_ode_kernel, grid, block, 0, stream,
                     s_grid, y0, W1, b1, W2, b2, G1, gb1, l1g, l1b,
                     G2, gb2, l2g, l2b, G3, gb3, out);
}
